// TimestepPredictionNetwork_86423331930278
// MI455X (gfx1250) — matrix operand order repaired and checked by reading, not compiled, not run on hardware
//
#include <hip/hip_runtime.h>
#include <hip/hip_bf16.h>
#include <math.h>

// ---------------------------------------------------------------------------
// ContinuousConv fluid network for MI455X (gfx1250, wave32, WMMA).
//
// Pipeline:
//   1) knn (K=48 fluid-fluid, K=32 box-fluid)  : LDS-tiled brute force + per-
//      thread max-heap in LDS (excludes d2<=0 like the reference).
//   2) geom : per (query,neighbor) pair -> poly6 window * ball->cube trilinear
//      weights (8 cells, 8 weights). Computed ONCE, reused by 4 fluid convs.
//   3) scatter : splat weighted features into a [16q x 64cell x 32ch] f32 LDS
//      accumulator (exclusive (q,ch) ownership -> no atomics), then emit f16
//      directly in the WMMA A-matrix lane/VGPR striping.
//   4) gemm_wmma : v_wmma_f32_16x16x32_f16, 2x2 register tile per wave
//      (4 independent accumulator chains, 32 B/lane global traffic per WMMA).
//      C operand carries the running f32 output so Cin-chunks / bias / dense
//      accumulate across sequential launches.
//   5) dense/relu/reduce : tiny VALU kernels.
// ---------------------------------------------------------------------------

#define NF 10000
#define NB 6000
#define KF 48
#define KB 32
#define MTILES (NF / 16)              // 625
#define RADIUS_F 0.1125f              // 1.5*6*0.025/2
#define EPSF 1e-12f

typedef _Float16 v16h __attribute__((ext_vector_type(16)));
typedef float    v8f  __attribute__((ext_vector_type(8)));

// ---------------------------------------------------------------- KNN -------
template <int K>
__global__ void knn_kernel(const float* __restrict__ qpos, int nq,
                           const float* __restrict__ ppos, int np,
                           int* __restrict__ out) {
  constexpr int BD = 64;   // threads per block (2 waves)
  constexpr int TS = 64;   // point tile size
  __shared__ float tpx[TS], tpy[TS], tpz[TS];
  __shared__ float hD[K * BD];
  __shared__ int   hI[K * BD];
  const int tid = threadIdx.x;
  const int q = blockIdx.x * BD + tid;
  const bool active = q < nq;
  float qx = 0.f, qy = 0.f, qz = 0.f;
  if (active) { qx = qpos[q*3+0]; qy = qpos[q*3+1]; qz = qpos[q*3+2]; }
  float* hd = &hD[tid];
  int*   hi = &hI[tid];
  int cnt = 0;
  for (int t0 = 0; t0 < np; t0 += TS) {
    __syncthreads();
    const int j = t0 + tid;
    if (j < np) { tpx[tid] = ppos[j*3+0]; tpy[tid] = ppos[j*3+1]; tpz[tid] = ppos[j*3+2]; }
    __syncthreads();
    const int lim = min(TS, np - t0);
    if (active) {
      for (int jj = 0; jj < lim; ++jj) {
        const float dx = qx - tpx[jj], dy = qy - tpy[jj], dz = qz - tpz[jj];
        const float d2 = dx*dx + dy*dy + dz*dz;
        if (d2 <= 0.f) continue;                     // radius_search_ignore_query_points
        if (cnt < K) {                               // heap push (sift up)
          int c = cnt++;
          while (c > 0) {
            const int p = (c - 1) >> 1;
            const float pd = hd[p*BD];
            if (pd < d2) { hd[c*BD] = pd; hi[c*BD] = hi[p*BD]; c = p; }
            else break;
          }
          hd[c*BD] = d2; hi[c*BD] = t0 + jj;
        } else if (d2 < hd[0]) {                     // replace root (sift down)
          int c = 0;
          for (;;) {
            const int l = 2*c + 1;
            if (l >= K) break;
            const int r = l + 1;
            const int m = (r < K && hd[r*BD] > hd[l*BD]) ? r : l;
            if (hd[m*BD] > d2) { hd[c*BD] = hd[m*BD]; hi[c*BD] = hi[m*BD]; c = m; }
            else break;
          }
          hd[c*BD] = d2; hi[c*BD] = t0 + jj;
        }
      }
    }
  }
  if (active) {
    for (int s = 0; s < K; ++s) out[(size_t)q*K + s] = (s < cnt) ? hi[s*BD] : 0;
  }
}

// ------------------------------------------------------------- geometry -----
__device__ __forceinline__ float sgnf(float a) {
  return (a > 0.f) ? 1.f : ((a < 0.f) ? -1.f : 0.f);
}

__global__ void geom_kernel(const float* __restrict__ qpos, const float* __restrict__ spos,
                            const int* __restrict__ idx, int nq, int K,
                            unsigned* __restrict__ cellsOut, float* __restrict__ wOut) {
  const int p = blockIdx.x * 256 + threadIdx.x;
  if (p >= nq * K) return;
  const int q = p / K;
  const int nb = idx[p];
  const float invr = 1.f / RADIUS_F;
  const float x = (spos[nb*3+0] - qpos[q*3+0]) * invr;
  const float y = (spos[nb*3+1] - qpos[q*3+1]) * invr;
  const float z = (spos[nb*3+2] - qpos[q*3+2]) * invr;
  const float r2 = x*x + y*y + z*z;
  const bool valid = (r2 <= 1.f) && (r2 > 0.f);
  unsigned clo = 0, chi = 0;
  float w8[8];
  if (valid) {
    float win = 1.f - r2; win = win * win * win;     // poly6 window
    const float r = sqrtf(r2);
    const float rho2 = x*x + y*y;
    const bool cap = (1.25f * z * z) > rho2;
    const float s = cap ? sqrtf(3.f * r / (r + fabsf(z) + EPSF))
                        : r / sqrtf(rho2 + EPSF);
    float xc = x * s, yc = y * s;
    float zc = cap ? (sgnf(z) * r) : (1.5f * z);
    if (!(r > EPSF)) { xc = 0.f; yc = 0.f; zc = 0.f; }
    const float rho = sqrtf(xc*xc + yc*yc);
    const bool xdom = fabsf(xc) >= fabsf(yc);
    const float FOPI = 1.2732395447351628f;          // 4/pi
    const float u1 = sgnf(xc) * rho;
    const float v1 = FOPI * u1 * atanf(yc / ((fabsf(xc) > EPSF) ? xc : 1.f));
    const float v2 = sgnf(yc) * rho;
    const float u2 = FOPI * v2 * atanf(xc / ((fabsf(yc) > EPSF) ? yc : 1.f));
    float u = 0.f, v = 0.f;
    if (rho > EPSF) { u = xdom ? u1 : u2; v = xdom ? v1 : v2; }
    const float g0 = (fminf(fmaxf(u,  -1.f), 1.f) + 1.f) * 1.5f;  // (KS-1)/2 = 1.5
    const float g1 = (fminf(fmaxf(v,  -1.f), 1.f) + 1.f) * 1.5f;
    const float g2 = (fminf(fmaxf(zc, -1.f), 1.f) + 1.f) * 1.5f;
    const float l0 = fminf(fmaxf(floorf(g0), 0.f), 2.f);
    const float l1 = fminf(fmaxf(floorf(g1), 0.f), 2.f);
    const float l2 = fminf(fmaxf(floorf(g2), 0.f), 2.f);
    const int i0 = (int)l0, i1 = (int)l1, i2 = (int)l2;
    const float f0 = g0 - l0, f1 = g1 - l1, f2 = g2 - l2;
#pragma unroll
    for (int j = 0; j < 8; ++j) {
      const int c0 = (j >> 2) & 1, c1 = (j >> 1) & 1, c2 = j & 1;
      const float w = (c0 ? f0 : 1.f - f0) * (c1 ? f1 : 1.f - f1) * (c2 ? f2 : 1.f - f2) * win;
      const unsigned cell = (unsigned)(((i0 + c0) * 4 + (i1 + c1)) * 4 + (i2 + c2));
      w8[j] = w;
      if (j < 4) clo |= cell << (j * 8);
      else       chi |= cell << ((j - 4) * 8);
    }
  } else {
#pragma unroll
    for (int j = 0; j < 8; ++j) w8[j] = 0.f;
  }
  cellsOut[(size_t)p*2 + 0] = clo;
  cellsOut[(size_t)p*2 + 1] = chi;
#pragma unroll
  for (int j = 0; j < 8; ++j) wOut[(size_t)p*8 + j] = w8[j];
}

// ---------------------------------------------------------------- misc ------
__global__ void fill_fl_kernel(const float* __restrict__ vel, float* __restrict__ fl) {
  const int n = blockIdx.x * 256 + threadIdx.x;
  if (n >= NF) return;
  fl[n*4+0] = 1.f;
  fl[n*4+1] = vel[n*3+0];
  fl[n*4+2] = vel[n*3+1];
  fl[n*4+3] = vel[n*3+2];
}

__global__ void relu_kernel(const float* __restrict__ in, float* __restrict__ out, int count) {
  const int i = blockIdx.x * 256 + threadIdx.x;
  if (i < count) out[i] = fmaxf(in[i], 0.f);
}

// out[n, coff+f] = bias[f] + (h? h[n]@Wd : 0) + (resid? resid[n,f] : 0)
__global__ void dense_kernel(const float* __restrict__ h, int cin,
                             const float* __restrict__ Wd, const float* __restrict__ bias,
                             const float* __restrict__ resid, int residStride,
                             float* __restrict__ out, int ostride, int coff, int cout, int n) {
  const int id = blockIdx.x * 256 + threadIdx.x;
  if (id >= n * cout) return;
  const int row = id / cout, f = id % cout;
  float s = bias[f];
  if (h) {
    const float* hr = h + (size_t)row * cin;
    for (int c = 0; c < cin; ++c) s += hr[c] * Wd[c * cout + f];
  }
  if (resid) s += resid[(size_t)row * residStride + f];
  out[(size_t)row * ostride + coff + f] = s;
}

// ------------------------------------------------------------- B packing ----
// WMMA B (32x16, f16) lane layout: lane = n + 16*((kk>>3)&1); pos = (kk&7) + 8*(kk>>4).
__global__ void packB_kernel(const float* __restrict__ W, int cin, int cout,
                             int KTB, int ntiles, _Float16* __restrict__ packedB) {
  const int id = blockIdx.x * 256 + threadIdx.x;
  if (id >= ntiles * KTB * 32) return;
  const int lane = id & 31;
  const int bt = (id >> 5) % KTB;
  const int nt = id / (32 * KTB);
  const int n  = lane & 15;
  const int hi = lane >> 4;
  const int Ktot = 64 * cin;
  _Float16* dst = packedB + (size_t)id * 16;
#pragma unroll
  for (int p = 0; p < 16; ++p) {
    const int kk = (p & 7) + 8 * hi + 16 * (p >> 3);
    const int Kg = bt * 32 + kk;
    const int colg = nt * 16 + n;
    const float v = (Kg < Ktot && colg < cout) ? W[(size_t)Kg * cout + colg] : 0.f;
    dst[p] = (_Float16)v;
  }
}

// ------------------------------------------------------------- scatter ------
// One block = 16 queries x one Cin chunk (<=32). Exclusive (q,c) ownership.
// Emits f16 in WMMA A (16x32) striping: lane = m + 16*((kk>>3)&1),
// pos = (kk&7) + 8*(kk>>4).
__global__ void scatter_kernel(const float* __restrict__ feat, int cinTotal, int cbase, int cchunk,
                               const int* __restrict__ idx, int K,
                               const unsigned* __restrict__ cells, const float* __restrict__ wgt,
                               _Float16* __restrict__ packedA, int KT) {
  __shared__ float acc[16 * 64 * 32];    // 128 KB (CDNA5 WGP has 320 KB LDS)
  const int mt = blockIdx.x;
  const int tid = threadIdx.x;
  const int npairs = 16 * cchunk;
  const int total = 16 * 64 * cchunk;
  for (int i = tid; i < total; i += 256) acc[i] = 0.f;
  __syncthreads();
  for (int p = tid; p < npairs; p += 256) {
    const int q = p / cchunk, c = p % cchunk;
    const int qg = mt * 16 + q;
    float* ab = &acc[(q * 64) * cchunk + c];
    for (int k = 0; k < K; ++k) {
      const int pr = qg * K + k;
      const int nb = idx[pr];
      const float f = feat[(size_t)nb * cinTotal + cbase + c];
      const unsigned clo = cells[(size_t)pr*2 + 0];
      const unsigned chi = cells[(size_t)pr*2 + 1];
      const float* w = &wgt[(size_t)pr * 8];
#pragma unroll
      for (int j = 0; j < 8; ++j) {
        const unsigned cell = (((j < 4) ? clo : chi) >> ((j & 3) * 8)) & 0xffu;
        ab[cell * cchunk] += w[j] * f;
      }
    }
  }
  __syncthreads();
  for (int p = tid; p < npairs; p += 256) {
    const int q = p / cchunk, c = p % cchunk;
    for (int cell = 0; cell < 64; ++cell) {
      const int lK = cell * cchunk + c;
      const int lkt = lK >> 5, kk = lK & 31;
      const int lane = q + 16 * ((kk >> 3) & 1);
      const int pos = (kk & 7) + 8 * (kk >> 4);
      packedA[((size_t)(mt * KT + lkt) * 32 + lane) * 16 + pos] =
          (_Float16)acc[(q * 64 + cell) * cchunk + c];
    }
  }
}

// ------------------------------------------------------------ WMMA GEMM -----
// 2x2 register tile per wave: 2 M-tiles x 2 N-tiles, 4 independent WMMA
// accumulator chains. Each k-step: 2 A + 2 B tile loads (128 B/lane) feed
// 4 WMMAs -> 32 B/lane per WMMA. C operand carries the running f32 output so
// sequential chunk launches accumulate. bt = lkt*ktStride + ktOff maps the
// chunk-local K tile onto the globally packed B tiles.
__global__ __launch_bounds__(32)
void gemm_wmma_kernel(const _Float16* __restrict__ A, int KT,
                      const _Float16* __restrict__ B, int KTB,
                      int ktStride, int ktOff, int mtiles, int ntiles,
                      float* __restrict__ out, int ostride, int coff, int coutValid) {
  const int npairs = (ntiles + 1) >> 1;
  const int mp = blockIdx.x / npairs;
  const int np = blockIdx.x % npairs;
  const int mt0 = mp * 2;
  const int mt1 = mt0 + 1;
  const bool m1v = mt1 < mtiles;
  const int nt0 = np * 2;
  const int nt1 = nt0 + 1;
  const bool n1v = nt1 < ntiles;
  const int lane = threadIdx.x;
  const int col = lane & 15;
  const int m0 = (lane >> 4) * 8;

  const int colg0 = nt0 * 16 + col;
  const int colg1 = nt1 * 16 + col;
  const bool v00 = colg0 < coutValid;
  const bool v01 = n1v && (colg1 < coutValid);
  const bool v10 = m1v && v00;
  const bool v11 = m1v && v01;
  const size_t r0 = (size_t)(mt0 * 16 + m0);
  const size_t r1 = (size_t)(mt1 * 16 + m0);

  v8f c00 = {}, c01 = {}, c10 = {}, c11 = {};
#pragma unroll
  for (int v = 0; v < 8; ++v) {
    if (v00) c00[v] = out[(r0 + v) * ostride + coff + colg0];
    if (v01) c01[v] = out[(r0 + v) * ostride + coff + colg1];
    if (v10) c10[v] = out[(r1 + v) * ostride + coff + colg0];
    if (v11) c11[v] = out[(r1 + v) * ostride + coff + colg1];
  }

  const _Float16* a0p = A + ((size_t)mt0 * KT * 32 + lane) * 16;
  const _Float16* a1p = A + ((size_t)(m1v ? mt1 : mt0) * KT * 32 + lane) * 16;
  const _Float16* b0p = B + ((size_t)nt0 * KTB * 32 + lane) * 16;
  const _Float16* b1p = B + ((size_t)(n1v ? nt1 : nt0) * KTB * 32 + lane) * 16;

#pragma unroll 2
  for (int lkt = 0; lkt < KT; ++lkt) {
    const v16h a0 = *(const v16h*)(a0p + (size_t)lkt * 512);
    const v16h a1 = *(const v16h*)(a1p + (size_t)lkt * 512);
    const size_t boff = (size_t)(lkt * ktStride + ktOff) * 512;
    const v16h b0 = *(const v16h*)(b0p + boff);
    const v16h b1 = *(const v16h*)(b1p + boff);
    if (lkt + 1 < KT)   // speculative prefetch of next A tile (global_prefetch_b8)
      __builtin_prefetch(a0p + (size_t)(lkt + 1) * 512, 0, 0);
    c00 = __builtin_amdgcn_wmma_f32_16x16x32_f16(false, a0, false, b0, (short)0, c00, false, false);
    c01 = __builtin_amdgcn_wmma_f32_16x16x32_f16(false, a0, false, b1, (short)0, c01, false, false);
    c10 = __builtin_amdgcn_wmma_f32_16x16x32_f16(false, a1, false, b0, (short)0, c10, false, false);
    c11 = __builtin_amdgcn_wmma_f32_16x16x32_f16(false, a1, false, b1, (short)0, c11, false, false);
  }

#pragma unroll
  for (int v = 0; v < 8; ++v) {
    if (v00) out[(r0 + v) * ostride + coff + colg0] = c00[v];
    if (v01) out[(r0 + v) * ostride + coff + colg1] = c01[v];
    if (v10) out[(r1 + v) * ostride + coff + colg0] = c10[v];
    if (v11) out[(r1 + v) * ostride + coff + colg1] = c11[v];
  }
}

// ------------------------------------------------------------- reduction ----
__global__ void reduce1_kernel(const float* __restrict__ in, int n, float* __restrict__ partial) {
  __shared__ float sm[256];
  float s = 0.f;
  for (int i = blockIdx.x * 256 + threadIdx.x; i < n; i += gridDim.x * 256) s += in[i];
  sm[threadIdx.x] = s;
  __syncthreads();
  for (int o = 128; o > 0; o >>= 1) {
    if (threadIdx.x < o) sm[threadIdx.x] += sm[threadIdx.x + o];
    __syncthreads();
  }
  if (threadIdx.x == 0) partial[blockIdx.x] = sm[0];
}

__global__ void reduce2_kernel(const float* __restrict__ partial, int n,
                               float* __restrict__ out, float scale) {
  __shared__ float sm[64];
  float s = 0.f;
  for (int i = threadIdx.x; i < n; i += 64) s += partial[i];
  sm[threadIdx.x] = s;
  __syncthreads();
  for (int o = 32; o > 0; o >>= 1) {
    if (threadIdx.x < o) sm[threadIdx.x] += sm[threadIdx.x + o];
    __syncthreads();
  }
  if (threadIdx.x == 0) out[0] = sm[0] * scale;
}

// ------------------------------------------------------------- launcher -----
extern "C" void kernel_launch(void* const* d_in, const int* in_sizes, int n_in,
                              void* d_out, int out_size, void* d_ws, size_t ws_size,
                              hipStream_t stream) {
  (void)in_sizes; (void)n_in; (void)out_size; (void)ws_size;
  const float* pos       = (const float*)d_in[0];
  const float* vel       = (const float*)d_in[1];
  const float* box       = (const float*)d_in[2];
  const float* box_feats = (const float*)d_in[3];
  const float* Wc0f = (const float*)d_in[4];  const float* bc0f = (const float*)d_in[5];
  const float* Wc0o = (const float*)d_in[6];  const float* bc0o = (const float*)d_in[7];
  const float* Wd0  = (const float*)d_in[8];  const float* bd0  = (const float*)d_in[9];
  const float* Wc1  = (const float*)d_in[10]; const float* bc1  = (const float*)d_in[11];
  const float* Wd1  = (const float*)d_in[12]; const float* bd1  = (const float*)d_in[13];
  const float* Wc2  = (const float*)d_in[14]; const float* bc2  = (const float*)d_in[15];
  const float* Wd2  = (const float*)d_in[16]; const float* bd2  = (const float*)d_in[17];
  const float* Wc3  = (const float*)d_in[18]; const float* bc3  = (const float*)d_in[19];
  const float* Wd3  = (const float*)d_in[20]; const float* bd3  = (const float*)d_in[21];

  // ---- workspace bump allocator (~85 MB total) ----
  char* wp = (char*)d_ws;
  auto alloc = [&](size_t bytes) -> void* {
    void* p = (void*)wp;
    wp += (bytes + 255) & ~(size_t)255;
    return p;
  };
  int*      idx_ff   = (int*)     alloc((size_t)NF * KF * 4);
  int*      idx_bf   = (int*)     alloc((size_t)NF * KB * 4);
  unsigned* cells_ff = (unsigned*)alloc((size_t)NF * KF * 8);
  float*    w_ff     = (float*)   alloc((size_t)NF * KF * 32);
  unsigned* cells_bf = (unsigned*)alloc((size_t)NF * KB * 8);
  float*    w_bf     = (float*)   alloc((size_t)NF * KB * 32);
  float*    fl       = (float*)   alloc((size_t)NF * 4 * 4);
  float*    bufA     = (float*)   alloc((size_t)NF * 96 * 4);  // layer0 out (96)
  float*    bufB     = (float*)   alloc((size_t)NF * 64 * 4);  // layer1 out
  float*    bufC     = (float*)   alloc((size_t)NF * 64 * 4);  // layer2 out
  float*    bufD     = (float*)   alloc((size_t)NF * 4);       // layer3 out (1)
  float*    hbuf     = (float*)   alloc((size_t)NF * 96 * 4);
  _Float16* pA       = (_Float16*)alloc((size_t)MTILES * 64 * 512 * 2);   // 41 MB
  _Float16* pB       = (_Float16*)alloc((size_t)4 * 192 * 512 * 2);       // 786 KB
  float*    partial  = (float*)   alloc(64 * 4);

  const int REDUCE_BLOCKS = 40;
  const int MPAIRS = (MTILES + 1) / 2;   // 313

  // ---- KNN + geometry (shared by all fluid convs) ----
  knn_kernel<KF><<<(NF + 63) / 64, 64, 0, stream>>>(pos, NF, pos, NF, idx_ff);
  knn_kernel<KB><<<(NF + 63) / 64, 64, 0, stream>>>(pos, NF, box, NB, idx_bf);
  geom_kernel<<<(NF * KF + 255) / 256, 256, 0, stream>>>(pos, pos, idx_ff, NF, KF, cells_ff, w_ff);
  geom_kernel<<<(NF * KB + 255) / 256, 256, 0, stream>>>(pos, box, idx_bf, NF, KB, cells_bf, w_bf);
  fill_fl_kernel<<<(NF + 255) / 256, 256, 0, stream>>>(vel, fl);

  // ---- layer 0: prev = concat(c0o[0:32], c0f[32:64], d0[64:96]) ----
  // c0f : cconv(fl, Cin=4 -> 32), K space = 256, KT = 8
  packB_kernel<<<(2*8*32 + 255)/256, 256, 0, stream>>>(Wc0f, 4, 32, 8, 2, pB);
  dense_kernel<<<(NF*32 + 255)/256, 256, 0, stream>>>(nullptr, 0, nullptr, bc0f, nullptr, 0,
                                                      bufA, 96, 32, 32, NF);
  scatter_kernel<<<MTILES, 256, 0, stream>>>(fl, 4, 0, 4, idx_ff, KF, cells_ff, w_ff, pA, 8);
  gemm_wmma_kernel<<<MPAIRS * 1, 32, 0, stream>>>(pA, 8, pB, 8, 1, 0, MTILES, 2,
                                                  bufA, 96, 32, 32);
  // c0o : cconv(box_feats, Cin=3 -> 32), K space = 192, KT = 6
  packB_kernel<<<(2*6*32 + 255)/256, 256, 0, stream>>>(Wc0o, 3, 32, 6, 2, pB);
  dense_kernel<<<(NF*32 + 255)/256, 256, 0, stream>>>(nullptr, 0, nullptr, bc0o, nullptr, 0,
                                                      bufA, 96, 0, 32, NF);
  scatter_kernel<<<MTILES, 256, 0, stream>>>(box_feats, 3, 0, 3, idx_bf, KB, cells_bf, w_bf, pA, 6);
  gemm_wmma_kernel<<<MPAIRS * 1, 32, 0, stream>>>(pA, 6, pB, 6, 1, 0, MTILES, 2,
                                                  bufA, 96, 0, 32);
  // d0 : fl(4) @ Wd0 + bd0 -> cols 64..95
  dense_kernel<<<(NF*32 + 255)/256, 256, 0, stream>>>(fl, 4, Wd0, bd0, nullptr, 0,
                                                      bufA, 96, 64, 32, NF);

  // ---- layer 1: h = relu(prev96); out64 = cconv(h,96->64) + h@Wd1 + bd1 ----
  relu_kernel<<<(NF*96 + 255)/256, 256, 0, stream>>>(bufA, hbuf, NF*96);
  dense_kernel<<<(NF*64 + 255)/256, 256, 0, stream>>>(hbuf, 96, Wd1, bd1, nullptr, 0,
                                                      bufB, 64, 0, 64, NF);
  packB_kernel<<<(4*192*32 + 255)/256, 256, 0, stream>>>(Wc1, 96, 64, 192, 4, pB);
  for (int cb = 0; cb < 96; cb += 32) {
    scatter_kernel<<<MTILES, 256, 0, stream>>>(hbuf, 96, cb, 32, idx_ff, KF, cells_ff, w_ff, pA, 64);
    gemm_wmma_kernel<<<MPAIRS * 2, 32, 0, stream>>>(pA, 64, pB, 192, 3, cb / 32, MTILES, 4,
                                                    bufB, 64, 0, 64);
  }

  // ---- layer 2: residual ----
  relu_kernel<<<(NF*64 + 255)/256, 256, 0, stream>>>(bufB, hbuf, NF*64);
  dense_kernel<<<(NF*64 + 255)/256, 256, 0, stream>>>(hbuf, 64, Wd2, bd2, bufB, 64,
                                                      bufC, 64, 0, 64, NF);
  packB_kernel<<<(4*128*32 + 255)/256, 256, 0, stream>>>(Wc2, 64, 64, 128, 4, pB);
  for (int cb = 0; cb < 64; cb += 32) {
    scatter_kernel<<<MTILES, 256, 0, stream>>>(hbuf, 64, cb, 32, idx_ff, KF, cells_ff, w_ff, pA, 64);
    gemm_wmma_kernel<<<MPAIRS * 2, 32, 0, stream>>>(pA, 64, pB, 128, 2, cb / 32, MTILES, 4,
                                                    bufC, 64, 0, 64);
  }

  // ---- layer 3: 64 -> 1 ----
  relu_kernel<<<(NF*64 + 255)/256, 256, 0, stream>>>(bufC, hbuf, NF*64);
  dense_kernel<<<(NF*1 + 255)/256, 256, 0, stream>>>(hbuf, 64, Wd3, bd3, nullptr, 0,
                                                     bufD, 1, 0, 1, NF);
  packB_kernel<<<(1*128*32 + 255)/256, 256, 0, stream>>>(Wc3, 64, 1, 128, 1, pB);
  for (int cb = 0; cb < 64; cb += 32) {
    scatter_kernel<<<MTILES, 256, 0, stream>>>(hbuf, 64, cb, 32, idx_ff, KF, cells_ff, w_ff, pA, 64);
    gemm_wmma_kernel<<<MPAIRS * 1, 32, 0, stream>>>(pA, 64, pB, 128, 2, cb / 32, MTILES, 1,
                                                    bufD, 1, 0, 1);
  }

  // ---- mean ----
  reduce1_kernel<<<REDUCE_BLOCKS, 256, 0, stream>>>(bufD, NF, partial);
  reduce2_kernel<<<1, 64, 0, stream>>>(partial, REDUCE_BLOCKS, (float*)d_out, 1.f / (float)NF);
}
